// GNNModel_9826885173265
// MI455X (gfx1250) — compile-verified, matrix operand order checked by reading, not run on hardware
//
#include <hip/hip_runtime.h>
#include <math.h>

typedef __attribute__((ext_vector_type(16))) _Float16 v16h;
typedef __attribute__((ext_vector_type(8)))  float    v8f;

#define Nn 50000
#define Ee 1600000
#define Cc 128
#define Gg 64
#define Ll 3
#define EPSf 1e-5f

// ---------------- device helpers ----------------

__device__ __forceinline__ float gelu_f(float v) {
    return 0.5f * v * (1.0f + erff(v * 0.70710678118654752f));
}
__device__ __forceinline__ float act_f(float v, int act) {
    if (act == 1) return v > 0.0f ? v : 0.01f * v;   // leaky_relu(0.01)
    if (act == 2) return gelu_f(v);                  // exact gelu
    return v;
}
// float atomic max via sign-split integer trick (works with -inf init)
__device__ __forceinline__ void atomicMaxF(float* addr, float val) {
    if (val >= 0.0f) atomicMax((int*)addr, __float_as_int(val));
    else             atomicMin((unsigned int*)addr, __float_as_uint(val));
}
__device__ __forceinline__ float wave_sum(float v) {
#pragma unroll
    for (int o = 16; o > 0; o >>= 1) v += __shfl_xor(v, o, 32);
    return v;
}

// ---------------- utility kernels ----------------

__global__ void fill_kernel(float* p, float v, int n) {
    for (int i = blockIdx.x * blockDim.x + threadIdx.x; i < n; i += gridDim.x * blockDim.x)
        p[i] = v;
}

__global__ void scatter_count_kernel(const int* __restrict__ idx, float* cnt, int n) {
    for (int i = blockIdx.x * blockDim.x + threadIdx.x; i < n; i += gridDim.x * blockDim.x)
        atomicAdd(&cnt[idx[i]], 1.0f);
}

// Repack row-major f32 K x C weight into WMMA B-fragment order (f16).
// Fragment (kb, nt): 32 lanes x 16 halfs contiguous.
// lane l: n = nt*16 + (l&15); half j: k = kb*32 + (l>=16?8:0) + (j&7) + ((j>>3)<<4)
__global__ void repack_kernel(const float* __restrict__ W, _Float16* __restrict__ Wp, int K, int C) {
    int NT = C >> 4;
    int total = K * C;
    for (int i = blockIdx.x * blockDim.x + threadIdx.x; i < total; i += gridDim.x * blockDim.x) {
        int j = i & 15;
        int lane = (i >> 4) & 31;
        int fn = i >> 9;
        int nt = fn % NT;
        int kb = fn / NT;
        int k = kb * 32 + ((lane >= 16) ? 8 : 0) + (j & 7) + ((j >> 3) << 4);
        int n = nt * 16 + (lane & 15);
        Wp[i] = (_Float16)W[k * C + n];
    }
}

// h = x @ W_in0 + b_in0  (N x 4 @ 4 x 128), trivial
__global__ void in_linear0_kernel(const float* __restrict__ x, const float* __restrict__ W,
                                  const float* __restrict__ b, float* __restrict__ h) {
    for (int i = blockIdx.x * blockDim.x + threadIdx.x; i < Nn * Cc; i += gridDim.x * blockDim.x) {
        int r = i >> 7, c = i & 127;
        float s = b[c];
#pragma unroll
        for (int k = 0; k < 4; ++k) s += x[r * 4 + k] * W[k * Cc + c];
        h[i] = s;
    }
}

// ---------------- GraphNorm (3 passes; batch==nullptr => whole-set, cnt=cntDflt) ----------------

__global__ void gn_sum_kernel(const float* __restrict__ X, const int* __restrict__ batch,
                              float* gsum, int n) {
    for (int i = blockIdx.x * blockDim.x + threadIdx.x; i < n; i += gridDim.x * blockDim.x) {
        int r = i >> 7, c = i & 127;
        int g = batch ? batch[r] : 0;
        atomicAdd(&gsum[g * Cc + c], X[i]);
    }
}

__global__ void gn_var_kernel(const float* __restrict__ X, const int* __restrict__ batch,
                              const float* __restrict__ gsum, const float* __restrict__ gcnt,
                              float cntDflt, const float* __restrict__ ms, float* gvsum, int n) {
    for (int i = blockIdx.x * blockDim.x + threadIdx.x; i < n; i += gridDim.x * blockDim.x) {
        int r = i >> 7, c = i & 127;
        int g = batch ? batch[r] : 0;
        float cnt = gcnt ? fmaxf(gcnt[g], 1.0f) : cntDflt;
        float mean = gsum[g * Cc + c] / cnt;
        float o = X[i] - mean * ms[c];
        atomicAdd(&gvsum[g * Cc + c], o * o);
    }
}

__global__ void gn_apply_kernel(const float* __restrict__ X, const int* __restrict__ batch,
                                const float* __restrict__ gsum, const float* __restrict__ gvsum,
                                const float* __restrict__ gcnt, float cntDflt,
                                const float* __restrict__ w, const float* __restrict__ b,
                                const float* __restrict__ ms, float* __restrict__ out,
                                int act, int n) {
    for (int i = blockIdx.x * blockDim.x + threadIdx.x; i < n; i += gridDim.x * blockDim.x) {
        int r = i >> 7, c = i & 127;
        int g = batch ? batch[r] : 0;
        float cnt = gcnt ? fmaxf(gcnt[g], 1.0f) : cntDflt;
        float mean = gsum[g * Cc + c] / cnt;
        float var = gvsum[g * Cc + c] / cnt;
        float o = X[i] - mean * ms[c];
        out[i] = act_f(w[c] * o * rsqrtf(var + EPSf) + b[c], act);
    }
}

// ---------------- generic node GEMM: Y[M x 128] = act(X[M x K] @ Wp + bias [+ residual]) ----------------

template <int KB>
__global__ __launch_bounds__(128) void node_gemm_kernel(
    const float* __restrict__ X, const _Float16* __restrict__ Wp,
    const float* __restrict__ bias, const float* __restrict__ residual,
    float* __restrict__ Y, int M, int act, int numTiles, int iters) {
    const int K = KB * 32;
    int lane = threadIdx.x & 31, wave = threadIdx.x >> 5;
    int gwave = blockIdx.x * 4 + wave, gwaves = gridDim.x * 4;
    int m = lane & 15;
    int hi8 = (lane >= 16) ? 8 : 0;
    for (int it = 0; it < iters; ++it) {
        int tile = it * gwaves + gwave;
        if (tile >= numTiles) continue;
        int row0 = tile * 16;
        int rowm = row0 + m; if (rowm > M - 1) rowm = M - 1;
        const float* xp = X + (size_t)rowm * K;
        v16h A[KB];
#pragma unroll
        for (int kb = 0; kb < KB; ++kb) {
            int base = kb * 32 + hi8;
            v16h a;
#pragma unroll
            for (int j = 0; j < 8; ++j) {
                a[j]     = (_Float16)xp[base + j];
                a[j + 8] = (_Float16)xp[base + 16 + j];
            }
            A[kb] = a;
        }
#pragma unroll
        for (int nt = 0; nt < 8; ++nt) {
            float bv = bias[nt * 16 + (lane & 15)];
            v8f acc = {bv, bv, bv, bv, bv, bv, bv, bv};
#pragma unroll
            for (int kb = 0; kb < KB; ++kb) {
                v16h bf = *(const v16h*)(Wp + (((size_t)kb * 8 + nt) * 32 + lane) * 16);
                acc = __builtin_amdgcn_wmma_f32_16x16x32_f16(false, A[kb], false, bf,
                                                             (short)0, acc, false, false);
            }
#pragma unroll
            for (int r = 0; r < 8; ++r) {
                int row = row0 + r + hi8;
                if (row < M) {
                    int col = nt * 16 + (lane & 15);
                    float v = acc[r];
                    if (residual) v += residual[(size_t)row * Cc + col];
                    Y[(size_t)row * Cc + col] = act_f(v, act);
                }
            }
        }
    }
}

// ---------------- EdgeConv message MLP (the hot kernel) ----------------
// per wave: 16 edges. A = [x_i | x_j - x_i] (16x256 f16, 8 K-blocks)
// GEMM1 (256->128) -> GELU -> LDS retile -> GEMM2 (128->128) -> atomic max/add by dst.

__global__ __launch_bounds__(128) void edge_mlp_kernel(
    const float* __restrict__ hn, const int* __restrict__ src, const int* __restrict__ dst,
    const _Float16* __restrict__ W0p, const float* __restrict__ b0,
    const _Float16* __restrict__ W1p, const float* __restrict__ b1,
    float* __restrict__ accMax, float* __restrict__ accSum,
    int numTiles, int iters) {
    __shared__ _Float16 midLds[4][16 * Cc];
    __shared__ int dstLds[4][16];
    int lane = threadIdx.x & 31, wave = threadIdx.x >> 5;
    int gwave = blockIdx.x * 4 + wave, gwaves = gridDim.x * 4;
    int m = lane & 15;
    bool hi = lane >= 16;
    int hi8 = hi ? 8 : 0;

    for (int it = 0; it < iters; ++it) {
        int tile = it * gwaves + gwave;           // uniform per wave
        bool valid = tile < numTiles;
        int e = valid ? (tile * 16 + m) : 0;
        int di = valid ? dst[e] : 0;
        int si = valid ? src[e] : 0;
        if (!hi) dstLds[wave][m] = di;
        const float* xi = hn + (size_t)di * Cc;
        const float* xj = hn + (size_t)si * Cc;

        // Build A fragments: kb 0..3 = x_i cols, kb 4..7 = (x_j - x_i) cols
        v16h A[8];
#pragma unroll
        for (int kb = 0; kb < 8; ++kb) {
            int base = (kb & 3) * 32 + hi8;
            v16h a;
            if (kb < 4) {
#pragma unroll
                for (int j = 0; j < 8; ++j) {
                    a[j]     = (_Float16)xi[base + j];
                    a[j + 8] = (_Float16)xi[base + 16 + j];
                }
            } else {
#pragma unroll
                for (int j = 0; j < 8; ++j) {
                    a[j]     = (_Float16)(xj[base + j]      - xi[base + j]);
                    a[j + 8] = (_Float16)(xj[base + 16 + j] - xi[base + 16 + j]);
                }
            }
            A[kb] = a;
        }

        // GEMM1 (16x256 @ 256x128) + GELU -> mid tile in LDS (A-layout for GEMM2)
#pragma unroll
        for (int nt = 0; nt < 8; ++nt) {
            float bv = b0[nt * 16 + (lane & 15)];
            v8f acc = {bv, bv, bv, bv, bv, bv, bv, bv};
#pragma unroll
            for (int kb = 0; kb < 8; ++kb) {
                v16h bf = *(const v16h*)(W0p + (((size_t)kb * 8 + nt) * 32 + lane) * 16);
                acc = __builtin_amdgcn_wmma_f32_16x16x32_f16(false, A[kb], false, bf,
                                                             (short)0, acc, false, false);
            }
#pragma unroll
            for (int r = 0; r < 8; ++r) {
                float v = gelu_f(acc[r]);
                midLds[wave][(r + hi8) * Cc + nt * 16 + (lane & 15)] = (_Float16)v;
            }
        }

        // wait in-wave LDS stores (cross-lane visibility); LDS is in-order per wave
        __asm__ volatile("s_wait_dscnt 0" ::: "memory");

        v16h A2[4];
#pragma unroll
        for (int kb = 0; kb < 4; ++kb) {
            int base = kb * 32 + hi8;
            v16h a;
#pragma unroll
            for (int j = 0; j < 8; ++j) {
                a[j]     = midLds[wave][m * Cc + base + j];
                a[j + 8] = midLds[wave][m * Cc + base + 16 + j];
            }
            A2[kb] = a;
        }
        int dstR[8];
#pragma unroll
        for (int r = 0; r < 8; ++r) dstR[r] = dstLds[wave][r + hi8];

        // GEMM2 (16x128 @ 128x128) + scatter-aggregate (segment max & sum by dst)
#pragma unroll
        for (int nt = 0; nt < 8; ++nt) {
            float bv = b1[nt * 16 + (lane & 15)];
            v8f acc = {bv, bv, bv, bv, bv, bv, bv, bv};
#pragma unroll
            for (int kb = 0; kb < 4; ++kb) {
                v16h bf = *(const v16h*)(W1p + (((size_t)kb * 8 + nt) * 32 + lane) * 16);
                acc = __builtin_amdgcn_wmma_f32_16x16x32_f16(false, A2[kb], false, bf,
                                                             (short)0, acc, false, false);
            }
            if (valid) {
#pragma unroll
                for (int r = 0; r < 8; ++r) {
                    float v = acc[r];
                    size_t off = (size_t)dstR[r] * Cc + nt * 16 + (lane & 15);
                    atomicAdd(accSum + off, v);
                    atomicMaxF(accMax + off, v);
                }
            }
        }
    }
}

// pooledE[i] = [ finite(max)?:0 , sum/deg ]
__global__ void combine_edge_kernel(const float* __restrict__ accMax, const float* __restrict__ accSum,
                                    const float* __restrict__ deg, float* __restrict__ pooledE, int n) {
    for (int i = blockIdx.x * blockDim.x + threadIdx.x; i < n; i += gridDim.x * blockDim.x) {
        int r = i >> 7, c = i & 127;
        float mx = accMax[i];
        if (!(fabsf(mx) <= 3.0e38f)) mx = 0.0f;
        pooledE[(size_t)r * 256 + c] = mx;
        pooledE[(size_t)r * 256 + 128 + c] = accSum[i] / fmaxf(deg[r], 1.0f);
    }
}

// ---------------- readout ----------------

__global__ void pool_atomic_kernel(const float* __restrict__ X, const int* __restrict__ batch,
                                   float* gmax, float* gsum, int n) {
    for (int i = blockIdx.x * blockDim.x + threadIdx.x; i < n; i += gridDim.x * blockDim.x) {
        int r = i >> 7, c = i & 127;
        int g = batch[r];
        float v = X[i];
        atomicMaxF(&gmax[g * Cc + c], v);
        atomicAdd(&gsum[g * Cc + c], v);
    }
}

__global__ void combine_pool_kernel(const float* __restrict__ gmax, const float* __restrict__ gsum,
                                    const float* __restrict__ gcnt, float* __restrict__ pooledG) {
    for (int i = blockIdx.x * blockDim.x + threadIdx.x; i < Gg * Cc; i += gridDim.x * blockDim.x) {
        int g = i >> 7, c = i & 127;
        float mx = gmax[i];
        if (!(fabsf(mx) <= 3.0e38f)) mx = 0.0f;
        pooledG[g * 256 + c] = mx;
        pooledG[g * 256 + 128 + c] = gsum[i] / fmaxf(gcnt[g], 1.0f);
    }
}

// one wave per row: LayerNorm over 128 + GELU
__global__ void ln_gelu_kernel(const float* __restrict__ P, const float* __restrict__ w,
                               const float* __restrict__ b, float* __restrict__ out, int rows) {
    int lane = threadIdx.x & 31, wave = threadIdx.x >> 5;
    int row = blockIdx.x * (blockDim.x >> 5) + wave;
    if (row >= rows) return;
    float v[4]; float s = 0.0f;
#pragma unroll
    for (int k = 0; k < 4; ++k) { v[k] = P[row * Cc + lane + 32 * k]; s += v[k]; }
    s = wave_sum(s);
    float mu = s * (1.0f / 128.0f);
    float q = 0.0f;
#pragma unroll
    for (int k = 0; k < 4; ++k) { float d = v[k] - mu; q += d * d; }
    q = wave_sum(q);
    float inv = rsqrtf(q * (1.0f / 128.0f) + EPSf);
#pragma unroll
    for (int k = 0; k < 4; ++k) {
        int c = lane + 32 * k;
        float o = w[c] * ((v[k] - mu) * inv) + b[c];
        out[row * Cc + c] = gelu_f(o);
    }
}

// one wave per graph: out[g] = p[g,:] . Wc1 + bc1
__global__ void final_linear_kernel(const float* __restrict__ P, const float* __restrict__ Wc1,
                                    const float* __restrict__ bc1, float* __restrict__ out, int rows) {
    int lane = threadIdx.x & 31, wave = threadIdx.x >> 5;
    int row = blockIdx.x * (blockDim.x >> 5) + wave;
    if (row >= rows) return;
    float s = 0.0f;
#pragma unroll
    for (int k = 0; k < 4; ++k) {
        int c = lane + 32 * k;
        s += P[row * Cc + c] * Wc1[c];
    }
    s = wave_sum(s);
    if (lane == 0) out[row] = s + bc1[0];
}

// ---------------- host ----------------

extern "C" void kernel_launch(void* const* d_in, const int* in_sizes, int n_in,
                              void* d_out, int out_size, void* d_ws, size_t ws_size,
                              hipStream_t stream) {
    (void)in_sizes; (void)n_in; (void)out_size; (void)ws_size;

    const float* x       = (const float*)d_in[0];
    const int*   ei      = (const int*)d_in[1];
    const int*   src     = ei;            // edge_index[0]
    const int*   dstp    = ei + Ee;       // edge_index[1]
    const int*   batch   = (const int*)d_in[2];
    const float* W_in0   = (const float*)d_in[4];
    const float* b_in0   = (const float*)d_in[5];
    const float* gn_in_w = (const float*)d_in[6];
    const float* gn_in_b = (const float*)d_in[7];
    const float* gn_in_ms= (const float*)d_in[8];
    const float* W_in1   = (const float*)d_in[9];
    const float* b_in1   = (const float*)d_in[10];
    const float* conv_W0 = (const float*)d_in[11];
    const float* conv_b0 = (const float*)d_in[12];
    const float* conv_W1 = (const float*)d_in[13];
    const float* conv_b1 = (const float*)d_in[14];
    const float* proj_W  = (const float*)d_in[15];
    const float* proj_b  = (const float*)d_in[16];
    const float* gn_w    = (const float*)d_in[17];
    const float* gn_b    = (const float*)d_in[18];
    const float* gn_ms   = (const float*)d_in[19];
    const float* fn_w    = (const float*)d_in[20];
    const float* fn_b    = (const float*)d_in[21];
    const float* fn_ms   = (const float*)d_in[22];
    const float* Wc0     = (const float*)d_in[23];
    const float* bc0     = (const float*)d_in[24];
    const float* ln_w    = (const float*)d_in[25];
    const float* ln_b    = (const float*)d_in[26];
    const float* Wc1     = (const float*)d_in[27];
    const float* bc1     = (const float*)d_in[28];
    float* outp = (float*)d_out;

    // workspace carve-out
    char* ws = (char*)d_ws;
    size_t off = 0;
    auto alloc = [&](size_t bytes) -> void* {
        void* p = ws + off;
        off += (bytes + 255) & ~(size_t)255;
        return p;
    };
    float* h       = (float*)alloc((size_t)Nn * Cc * 4);
    float* hnb     = (float*)alloc((size_t)Nn * Cc * 4);
    float* accMax  = (float*)alloc((size_t)Nn * Cc * 4);
    float* accSum  = (float*)alloc((size_t)Nn * Cc * 4);
    float* pooledE = (float*)alloc((size_t)Nn * 256 * 4);
    float* deg     = (float*)alloc((size_t)Nn * 4);
    float* gcnt    = (float*)alloc((size_t)Gg * 4);
    float* gsum    = (float*)alloc((size_t)Gg * Cc * 4);
    float* gvsum   = (float*)alloc((size_t)Gg * Cc * 4);
    float* gmaxP   = (float*)alloc((size_t)Gg * Cc * 4);
    float* gsumP   = (float*)alloc((size_t)Gg * Cc * 4);
    float* pooledG = (float*)alloc((size_t)Gg * 256 * 4);
    float* pbuf    = (float*)alloc((size_t)Gg * Cc * 4);
    float* pbuf2   = (float*)alloc((size_t)Gg * Cc * 4);
    _Float16* Win1p = (_Float16*)alloc((size_t)128 * 128 * 2);
    _Float16* W0p   = (_Float16*)alloc((size_t)Ll * 256 * 128 * 2);
    _Float16* W1p   = (_Float16*)alloc((size_t)Ll * 128 * 128 * 2);
    _Float16* Pp    = (_Float16*)alloc((size_t)Ll * 256 * 128 * 2);
    _Float16* Wc0p  = (_Float16*)alloc((size_t)256 * 128 * 2);

    const int nNC = Nn * Cc;
    const int ewBlocks = (nNC + 255) / 256;
    const float NEG_INF = -INFINITY;

    // ---- weight repack (f32 -> WMMA f16 fragment order) ----
    repack_kernel<<<(128 * 128 + 255) / 256, 256, 0, stream>>>(W_in1, Win1p, 128, 128);
    for (int l = 0; l < Ll; ++l) {
        repack_kernel<<<(256 * 128 + 255) / 256, 256, 0, stream>>>(conv_W0 + (size_t)l * 256 * 128, W0p + (size_t)l * 256 * 128, 256, 128);
        repack_kernel<<<(128 * 128 + 255) / 256, 256, 0, stream>>>(conv_W1 + (size_t)l * 128 * 128, W1p + (size_t)l * 128 * 128, 128, 128);
        repack_kernel<<<(256 * 128 + 255) / 256, 256, 0, stream>>>(proj_W + (size_t)l * 256 * 128, Pp + (size_t)l * 256 * 128, 256, 128);
    }
    repack_kernel<<<(256 * 128 + 255) / 256, 256, 0, stream>>>(Wc0, Wc0p, 256, 128);

    // ---- degrees & graph counts ----
    fill_kernel<<<(Nn + 255) / 256, 256, 0, stream>>>(deg, 0.0f, Nn);
    scatter_count_kernel<<<(Ee + 255) / 256, 256, 0, stream>>>(dstp, deg, Ee);
    fill_kernel<<<1, 64, 0, stream>>>(gcnt, 0.0f, Gg);
    scatter_count_kernel<<<(Nn + 255) / 256, 256, 0, stream>>>(batch, gcnt, Nn);

    // ---- input net ----
    in_linear0_kernel<<<ewBlocks, 256, 0, stream>>>(x, W_in0, b_in0, h);
    // whole-set GraphNorm + GELU
    fill_kernel<<<(Cc + 255) / 256, 256, 0, stream>>>(gsum, 0.0f, Cc);
    gn_sum_kernel<<<ewBlocks, 256, 0, stream>>>(h, nullptr, gsum, nNC);
    fill_kernel<<<(Cc + 255) / 256, 256, 0, stream>>>(gvsum, 0.0f, Cc);
    gn_var_kernel<<<ewBlocks, 256, 0, stream>>>(h, nullptr, gsum, nullptr, (float)Nn, gn_in_ms, gvsum, nNC);
    gn_apply_kernel<<<ewBlocks, 256, 0, stream>>>(h, nullptr, gsum, gvsum, nullptr, (float)Nn,
                                                  gn_in_w, gn_in_b, gn_in_ms, hnb, 2, nNC);
    // h = hn @ W_in1 + b_in1  (WMMA)
    {
        int tiles = Nn / 16, blocks = 512, gw = blocks * 4;
        int iters = (tiles + gw - 1) / gw;
        node_gemm_kernel<4><<<blocks, 128, 0, stream>>>(hnb, Win1p, b_in1, nullptr, h, Nn, 0, tiles, iters);
    }

    // ---- EdgeConv layers ----
    for (int l = 0; l < Ll; ++l) {
        // per-graph GraphNorm + leaky_relu -> hn
        fill_kernel<<<(Gg * Cc + 255) / 256, 256, 0, stream>>>(gsum, 0.0f, Gg * Cc);
        gn_sum_kernel<<<ewBlocks, 256, 0, stream>>>(h, batch, gsum, nNC);
        fill_kernel<<<(Gg * Cc + 255) / 256, 256, 0, stream>>>(gvsum, 0.0f, Gg * Cc);
        gn_var_kernel<<<ewBlocks, 256, 0, stream>>>(h, batch, gsum, gcnt, 1.0f, gn_ms + l * Cc, gvsum, nNC);
        gn_apply_kernel<<<ewBlocks, 256, 0, stream>>>(h, batch, gsum, gvsum, gcnt, 1.0f,
                                                      gn_w + l * Cc, gn_b + l * Cc, gn_ms + l * Cc,
                                                      hnb, 1, nNC);
        // aggregation accumulators
        fill_kernel<<<ewBlocks, 256, 0, stream>>>(accMax, NEG_INF, nNC);
        fill_kernel<<<ewBlocks, 256, 0, stream>>>(accSum, 0.0f, nNC);
        // edge MLP + scatter
        {
            int tiles = Ee / 16, blocks = 1024, gw = blocks * 4;
            int iters = (tiles + gw - 1) / gw;
            edge_mlp_kernel<<<blocks, 128, 0, stream>>>(
                hnb, src, dstp,
                W0p + (size_t)l * 256 * 128, conv_b0 + l * Cc,
                W1p + (size_t)l * 128 * 128, conv_b1 + l * Cc,
                accMax, accSum, tiles, iters);
        }
        combine_edge_kernel<<<ewBlocks, 256, 0, stream>>>(accMax, accSum, deg, pooledE, nNC);
        // h = pooledE @ proj_W + proj_b + identity
        {
            int tiles = Nn / 16, blocks = 512, gw = blocks * 4;
            int iters = (tiles + gw - 1) / gw;
            node_gemm_kernel<8><<<blocks, 128, 0, stream>>>(pooledE, Pp + (size_t)l * 256 * 128,
                                                            proj_b + l * Cc, h, h, Nn, 0, tiles, iters);
        }
    }

    // ---- final GraphNorm + leaky ----
    fill_kernel<<<(Gg * Cc + 255) / 256, 256, 0, stream>>>(gsum, 0.0f, Gg * Cc);
    gn_sum_kernel<<<ewBlocks, 256, 0, stream>>>(h, batch, gsum, nNC);
    fill_kernel<<<(Gg * Cc + 255) / 256, 256, 0, stream>>>(gvsum, 0.0f, Gg * Cc);
    gn_var_kernel<<<ewBlocks, 256, 0, stream>>>(h, batch, gsum, gcnt, 1.0f, fn_ms, gvsum, nNC);
    gn_apply_kernel<<<ewBlocks, 256, 0, stream>>>(h, batch, gsum, gvsum, gcnt, 1.0f,
                                                  fn_w, fn_b, fn_ms, hnb, 1, nNC);

    // ---- per-graph pooling ----
    fill_kernel<<<(Gg * Cc + 255) / 256, 256, 0, stream>>>(gmaxP, NEG_INF, Gg * Cc);
    fill_kernel<<<(Gg * Cc + 255) / 256, 256, 0, stream>>>(gsumP, 0.0f, Gg * Cc);
    pool_atomic_kernel<<<ewBlocks, 256, 0, stream>>>(hnb, batch, gmaxP, gsumP, nNC);
    combine_pool_kernel<<<(Gg * Cc + 255) / 256, 256, 0, stream>>>(gmaxP, gsumP, gcnt, pooledG);

    // ---- classifier ----
    {
        int tiles = Gg / 16, blocks = 1, gw = 4;
        int iters = (tiles + gw - 1) / gw;
        node_gemm_kernel<8><<<blocks, 128, 0, stream>>>(pooledG, Wc0p, bc0, nullptr, pbuf, Gg, 0, tiles, iters);
    }
    ln_gelu_kernel<<<Gg / 4, 128, 0, stream>>>(pbuf, ln_w, ln_b, pbuf2, Gg);
    final_linear_kernel<<<Gg / 4, 128, 0, stream>>>(pbuf2, Wc1, bc1, outp, Gg);
}
